// HierarchicalCrossAttention_72662256714490
// MI455X (gfx1250) — matrix operand order repaired and checked by reading, not compiled, not run on hardware
//
#include <hip/hip_runtime.h>
#include <hip/hip_fp16.h>
#include <math.h>

typedef _Float16 half_t;
typedef __attribute__((ext_vector_type(16))) _Float16 v16h;
typedef __attribute__((ext_vector_type(8)))  _Float16 v8h;
typedef __attribute__((ext_vector_type(4)))  _Float16 v4h;
typedef __attribute__((ext_vector_type(8)))  float    v8f;

#define T_TF   9
#define D_DIM  128
#define B_BATCH 16384
#define TM     64          // batch rows per block in fused kernel

// ---------------------------------------------------------------------------
// WMMA fragment loaders (wave32, v_wmma_f32_16x16x32_f16 layouts per ISA 7.12.2)
// ---------------------------------------------------------------------------
__device__ __forceinline__ v16h frag_a(const half_t* __restrict__ p, int row0,
                                       int kbase, int ld) {
  int lane = threadIdx.x & 31;
  int r = row0 + (lane & 15);
  int k = kbase + ((lane >> 4) << 3);
  const half_t* base = p + r * ld + k;
  v8h lo = *(const v8h*)(base);
  v8h hi = *(const v8h*)(base + 16);
  v16h a;
#pragma unroll
  for (int i = 0; i < 8; ++i) { a[i] = lo[i]; a[i + 8] = hi[i]; }
  return a;
}

__device__ __forceinline__ v16h frag_b(const half_t* __restrict__ p, int ncol,
                                       int kbase, int ld) {
  int lane = threadIdx.x & 31;
  int n = ncol + (lane & 15);
  int k = kbase + ((lane >> 4) << 3);
  const half_t* base = p + n * ld + k;
  v8h lo = *(const v8h*)(base);
  v8h hi = *(const v8h*)(base + 16);
  v16h b;
#pragma unroll
  for (int i = 0; i < 8; ++i) { b[i] = lo[i]; b[i + 8] = hi[i]; }
  return b;
}

__device__ __forceinline__ v8f vzero8() {
  v8f z;
#pragma unroll
  for (int i = 0; i < 8; ++i) z[i] = 0.0f;
  return z;
}

// gfx1250 async global->LDS copy (ASYNCcnt-tracked); lds_off = LDS byte offset
__device__ __forceinline__ void async_b128(unsigned int lds_off,
                                           unsigned long long gaddr) {
  asm volatile("global_load_async_to_lds_b128 %0, %1, off"
               :: "v"(lds_off), "v"(gaddr) : "memory");
}
__device__ __forceinline__ void wait_async0() {
  asm volatile("s_wait_asynccnt 0" ::: "memory");
}

// ---------------------------------------------------------------------------
// Kernel: convert x (f32) -> x16 (f16), same layout [T,B,D]
// ---------------------------------------------------------------------------
__global__ void k_prep_x(const float* __restrict__ x, half_t* __restrict__ x16,
                         int ntot) {
  int i = (blockIdx.x * 256 + threadIdx.x) * 4;
  if (i + 3 < ntot) {
    float4 v = *(const float4*)(x + i);
    v4h h;
    h[0] = (half_t)v.x; h[1] = (half_t)v.y; h[2] = (half_t)v.z; h[3] = (half_t)v.w;
    *(v4h*)(x16 + i) = h;
  }
}

// ---------------------------------------------------------------------------
// Kernel: per connection c, Wvo = Wv[c] @ Wo[c]  (folded V->O projection),
// stored TRANSPOSED in f16 as WvoT[c][j][i] for B-fragment loads.
// Also bvo[c][j] = bv[c] @ Wo[c] + bo[c].  Uses WMMA for the 128^3 fold.
// ---------------------------------------------------------------------------
__global__ void __launch_bounds__(256, 1)
k_prep_wvo(const float* __restrict__ Wv, const float* __restrict__ Wo,
           const float* __restrict__ bv, const float* __restrict__ bo,
           half_t* __restrict__ WvoT, float* __restrict__ bvo) {
  __shared__ __align__(32) unsigned char smem[65536];
  half_t* Av  = (half_t*)smem;              // 128x128 f16, Wv row-major
  half_t* BoT = (half_t*)(smem + 32768);    // 128x128 f16, Wo^T ([j][d])

  int c = blockIdx.x;
  int tid = threadIdx.x, wave = tid >> 5, lane = tid & 31;
  int ncol = wave * 16;
  const float* wv = Wv + (size_t)c * 16384;
  const float* wo = Wo + (size_t)c * 16384;

  for (int i = tid; i < 16384; i += 256) Av[i] = (half_t)wv[i];
  for (int i = tid; i < 16384; i += 256) {
    int j = i >> 7, dd = i & 127;
    BoT[j * 128 + dd] = (half_t)wo[dd * 128 + j];
  }
  __syncthreads();

  v8f acc[8];
#pragma unroll
  for (int mt = 0; mt < 8; ++mt) acc[mt] = vzero8();

#pragma unroll
  for (int ks = 0; ks < 4; ++ks) {
    v16h b = frag_b(BoT, ncol, ks * 32, 128);
#pragma unroll
    for (int mt = 0; mt < 8; ++mt) {
      v16h a = frag_a(Av, mt * 16, ks * 32, 128);
      acc[mt] = __builtin_amdgcn_wmma_f32_16x16x32_f16(false, a, false, b,
                                                       (short)0, acc[mt],
                                                       false, false);
    }
  }

  int n = ncol + (lane & 15);
  half_t* out = WvoT + (size_t)c * 16384;
#pragma unroll
  for (int mt = 0; mt < 8; ++mt)
#pragma unroll
    for (int r = 0; r < 8; ++r) {
      int m = mt * 16 + r + ((lane >> 4) << 3);
      out[n * 128 + m] = (half_t)acc[mt][r];   // transposed store: [j][i]
    }

  if (tid < 128) {
    int j = tid;
    float s = bo[c * 128 + j];
    for (int d = 0; d < 128; ++d) s += bv[c * 128 + d] * wo[d * 128 + j];
    bvo[c * 128 + j] = s;
  }
}

// ---------------------------------------------------------------------------
// Kernel: transpose+convert Wg[t] (256x128) -> WgT[t][n][k] f16 (k=0..255)
//         and Wp[t] (128x128) -> WpT[t][n][k] f16
// ---------------------------------------------------------------------------
__global__ void k_prep_wgwp(const float* __restrict__ Wg, const float* __restrict__ Wp,
                            half_t* __restrict__ WgT, half_t* __restrict__ WpT) {
  int t = blockIdx.x, tid = threadIdx.x;
  const float* wg = Wg + (size_t)t * 256 * 128;
  half_t* og = WgT + (size_t)t * 128 * 256;
  for (int i = tid; i < 32768; i += 256) {
    int n = i >> 8, k = i & 255;
    og[i] = (half_t)wg[k * 128 + n];
  }
  const float* wp = Wp + (size_t)t * 16384;
  half_t* op = WpT + (size_t)t * 16384;
  for (int i = tid; i < 16384; i += 256) {
    int n = i >> 7, k = i & 127;
    op[i] = (half_t)wp[k * 128 + n];
  }
}

// ---------------------------------------------------------------------------
// Fused kernel: per (timeframe t, 64-row batch tile):
//   cross_avg -> gate -> fused -> proj -> residual -> LN -> LN -> out
// 8 waves; wave w owns output columns [16w,16w+16).
// Stage 1 is double-buffered with GLOBAL_LOAD_ASYNC_TO_LDS_B128.
// LDS: A0 16K | A1 16K | B0 32K | B1 32K | Cbuf 16K = 112K
//      (B1 aliased as f32 Ybuf for the LN stage)
// ---------------------------------------------------------------------------
__global__ void __launch_bounds__(256, 1)
k_fused(const float* __restrict__ x, const half_t* __restrict__ x16,
        const half_t* __restrict__ WvoT, const float* __restrict__ bvo,
        const half_t* __restrict__ WgT, const float* __restrict__ bg,
        const half_t* __restrict__ WpT, const float* __restrict__ bp,
        const float* __restrict__ g1, const float* __restrict__ b1,
        const float* __restrict__ g2, const float* __restrict__ b2,
        const int* __restrict__ srcI, const int* __restrict__ tgtI,
        int C, float* __restrict__ out) {
  __shared__ __align__(32) unsigned char smem[114688];
  __shared__ int connList[64];
  __shared__ int connCnt;
  half_t* A0   = (half_t*)smem;               // 64x128 f16
  half_t* A1   = (half_t*)(smem + 16384);     // 64x128 f16
  half_t* B0   = (half_t*)(smem + 32768);     // 128x128 f16
  half_t* B1   = (half_t*)(smem + 65536);     // 128x128 f16
  half_t* Cbuf = (half_t*)(smem + 98304);     // 64x128 f16 (ca, then fused)
  float*  Ybuf = (float*)(smem + 65536);      // alias B1: 64x128 f32

  int t = blockIdx.y;
  int row0 = blockIdx.x * TM;
  int tid = threadIdx.x, wave = tid >> 5, lane = tid & 31;
  int ncol = wave * 16;
  int n = ncol + (lane & 15);

  if (tid == 0) {
    int k = 0;
    for (int c = 0; c < C; ++c)
      if (srcI[c] == t) connList[k++] = c;
    connCnt = k;
  }
  __syncthreads();
  int ncon = connCnt;

  // LDS byte offsets of the double buffers (low 32 bits of shared pointers)
  unsigned int aOff[2] = { (unsigned int)(uintptr_t)A0, (unsigned int)(uintptr_t)A1 };
  unsigned int bOff[2] = { (unsigned int)(uintptr_t)B0, (unsigned int)(uintptr_t)B1 };
  unsigned int slice = wave * 2048 + lane * 16;   // this lane's slice of A copies
  unsigned int sliceB = wave * 4096 + lane * 16;  // this lane's slice of B copies

  // ---- stage 1: cross accumulation, async double-buffered ----
  v8f acc[4];
#pragma unroll
  for (int mt = 0; mt < 4; ++mt) acc[mt] = vzero8();

  auto issue_conn = [&](int idx, int p) {
    int c = connList[idx];
    int tg = tgtI[c];
    unsigned long long gA =
        (unsigned long long)(x16 + ((size_t)tg * B_BATCH + row0) * 128) + slice;
    unsigned long long gB =
        (unsigned long long)(WvoT + (size_t)c * 16384) + sliceB;
    unsigned int la = aOff[p] + slice;
    unsigned int lb = bOff[p] + sliceB;
#pragma unroll
    for (int it = 0; it < 4; ++it)            // 16KB A tile: 8 waves x 4 x 512B
      async_b128(la + it * 512, gA + it * 512);
#pragma unroll
    for (int it = 0; it < 8; ++it)            // 32KB B tile: 8 waves x 8 x 512B
      async_b128(lb + it * 512, gB + it * 512);
  };

  if (ncon > 0) issue_conn(0, 0);
  for (int i = 0; i < ncon; ++i) {
    int p = i & 1;
    wait_async0();          // own slice of buffer p landed
    __syncthreads();        // everyone's slice landed; prev GEMM reads done
    if (i + 1 < ncon) issue_conn(i + 1, 1 - p);
    const half_t* Ab = p ? A1 : A0;
    const half_t* Bb = p ? B1 : B0;
#pragma unroll
    for (int ks = 0; ks < 4; ++ks) {
      v16h b = frag_b(Bb, ncol, ks * 32, 128);
#pragma unroll
      for (int mt = 0; mt < 4; ++mt) {
        v16h a = frag_a(Ab, mt * 16, ks * 32, 128);
        acc[mt] = __builtin_amdgcn_wmma_f32_16x16x32_f16(false, a, false, b,
                                                         (short)0, acc[mt],
                                                         false, false);
      }
    }
  }

  float invc = ncon > 0 ? 1.0f / (float)ncon : 0.0f;
  float bav = 0.0f;
  for (int c = 0; c < C; ++c)
    if (srcI[c] == t) bav += bvo[c * 128 + n];
  bav *= invc;

  __syncthreads();
  // ca -> Cbuf (f16), refill A0 with own x tile, B0 <- WgT half 0 (x part)
#pragma unroll
  for (int mt = 0; mt < 4; ++mt)
#pragma unroll
    for (int r = 0; r < 8; ++r) {
      int m = mt * 16 + r + ((lane >> 4) << 3);
      Cbuf[m * 128 + n] = (half_t)(acc[mt][r] * invc + bav);
    }
  {
    const uint4* s = (const uint4*)(x16 + ((size_t)t * B_BATCH + row0) * 128);
    uint4* d = (uint4*)A0;
    for (int i = tid; i < 1024; i += 256) d[i] = s[i];
    const half_t* wg = WgT + (size_t)t * 128 * 256;
    uint4* d2 = (uint4*)B0;
    for (int i = tid; i < 2048; i += 256) {
      int nn = i >> 4, k8 = i & 15;
      d2[nn * 16 + k8] = *(const uint4*)(wg + nn * 256 + k8 * 8);
    }
  }
  __syncthreads();

  // ---- stage 2: gate = sigmoid([x, ca] @ Wg + bg) ----
  v8f gacc[4];
#pragma unroll
  for (int mt = 0; mt < 4; ++mt) gacc[mt] = vzero8();
#pragma unroll
  for (int ks = 0; ks < 4; ++ks) {               // k = 0..127 : x part
    v16h b = frag_b(B0, ncol, ks * 32, 128);
#pragma unroll
    for (int mt = 0; mt < 4; ++mt) {
      v16h a = frag_a(A0, mt * 16, ks * 32, 128);
      gacc[mt] = __builtin_amdgcn_wmma_f32_16x16x32_f16(false, a, false, b,
                                                        (short)0, gacc[mt],
                                                        false, false);
    }
  }
  __syncthreads();
  {
    const half_t* wg = WgT + (size_t)t * 128 * 256;
    uint4* d2 = (uint4*)B0;
    for (int i = tid; i < 2048; i += 256) {
      int nn = i >> 4, k8 = i & 15;
      d2[nn * 16 + k8] = *(const uint4*)(wg + nn * 256 + 128 + k8 * 8);
    }
  }
  __syncthreads();
#pragma unroll
  for (int ks = 0; ks < 4; ++ks) {               // k = 128..255 : cross_avg part
    v16h b = frag_b(B0, ncol, ks * 32, 128);
#pragma unroll
    for (int mt = 0; mt < 4; ++mt) {
      v16h a = frag_a(Cbuf, mt * 16, ks * 32, 128);
      gacc[mt] = __builtin_amdgcn_wmma_f32_16x16x32_f16(false, a, false, b,
                                                        (short)0, gacc[mt],
                                                        false, false);
    }
  }
  __syncthreads();

  // ---- stage 3: fused = g*ca + (1-g)*x  (overwrite Cbuf) ----
  float bgn = bg[t * 128 + n];
#pragma unroll
  for (int mt = 0; mt < 4; ++mt)
#pragma unroll
    for (int r = 0; r < 8; ++r) {
      int m = mt * 16 + r + ((lane >> 4) << 3);
      float g = 1.0f / (1.0f + expf(-(gacc[mt][r] + bgn)));
      float cav = (float)Cbuf[m * 128 + n];
      float xv = x[((size_t)t * B_BATCH + row0 + m) * 128 + n];
      Cbuf[m * 128 + n] = (half_t)(g * cav + (1.0f - g) * xv);
    }
  __syncthreads();
  {
    const uint4* s2 = (const uint4*)(WpT + (size_t)t * 16384);
    uint4* d2 = (uint4*)B0;
    for (int i = tid; i < 2048; i += 256) d2[i] = s2[i];
  }
  __syncthreads();

  // ---- stage 4: proj = fused @ Wp + bp ----
  v8f pacc[4];
#pragma unroll
  for (int mt = 0; mt < 4; ++mt) pacc[mt] = vzero8();
#pragma unroll
  for (int ks = 0; ks < 4; ++ks) {
    v16h b = frag_b(B0, ncol, ks * 32, 128);
#pragma unroll
    for (int mt = 0; mt < 4; ++mt) {
      v16h a = frag_a(Cbuf, mt * 16, ks * 32, 128);
      pacc[mt] = __builtin_amdgcn_wmma_f32_16x16x32_f16(false, a, false, b,
                                                        (short)0, pacc[mt],
                                                        false, false);
    }
  }
  __syncthreads();   // B1 now reused as f32 Ybuf

  // ---- stage 5: y = x + proj + bp -> Ybuf (f32) ----
  float bpn = bp[t * 128 + n];
#pragma unroll
  for (int mt = 0; mt < 4; ++mt)
#pragma unroll
    for (int r = 0; r < 8; ++r) {
      int m = mt * 16 + r + ((lane >> 4) << 3);
      float xv = x[((size_t)t * B_BATCH + row0 + m) * 128 + n];
      Ybuf[m * 128 + n] = xv + pacc[mt][r] + bpn;
    }
  __syncthreads();

  // ---- stage 6: double LayerNorm, row-wise over D=128; 4 lanes per row ----
  {
    int row = tid >> 2, q = tid & 3;
    const float* yr = Ybuf + row * 128 + q * 32;
    float s = 0.0f, ss = 0.0f;
    float z[32];
#pragma unroll
    for (int j = 0; j < 32; ++j) { float v = yr[j]; s += v; ss += v * v; }
    s  += __shfl_xor(s, 1, 32);  s  += __shfl_xor(s, 2, 32);
    ss += __shfl_xor(ss, 1, 32); ss += __shfl_xor(ss, 2, 32);
    float mu = s * (1.0f / 128.0f);
    float var = ss * (1.0f / 128.0f) - mu * mu;
    float rs = rsqrtf(var + 1e-5f);
    const float* g1t = g1 + t * 128 + q * 32;
    const float* b1t = b1 + t * 128 + q * 32;
    float s2 = 0.0f, ss2 = 0.0f;
#pragma unroll
    for (int j = 0; j < 32; ++j) {
      float zz = (yr[j] - mu) * rs * g1t[j] + b1t[j];
      z[j] = zz; s2 += zz; ss2 += zz * zz;
    }
    s2  += __shfl_xor(s2, 1, 32);  s2  += __shfl_xor(s2, 2, 32);
    ss2 += __shfl_xor(ss2, 1, 32); ss2 += __shfl_xor(ss2, 2, 32);
    float mu2 = s2 * (1.0f / 128.0f);
    float var2 = ss2 * (1.0f / 128.0f) - mu2 * mu2;
    float rs2 = rsqrtf(var2 + 1e-5f);
    const float* g2t = g2 + t * 128 + q * 32;
    const float* b2t = b2 + t * 128 + q * 32;
    float* o = out + ((size_t)t * B_BATCH + row0 + row) * 128 + q * 32;
#pragma unroll
    for (int j = 0; j < 32; ++j)
      o[j] = (z[j] - mu2) * rs2 * g2t[j] + b2t[j];
  }
}

// ---------------------------------------------------------------------------
extern "C" void kernel_launch(void* const* d_in, const int* in_sizes, int n_in,
                              void* d_out, int out_size, void* d_ws, size_t ws_size,
                              hipStream_t stream) {
  const float* x  = (const float*)d_in[0];
  const float* Wv = (const float*)d_in[5];
  const float* bv = (const float*)d_in[6];
  const float* Wo = (const float*)d_in[7];
  const float* bo = (const float*)d_in[8];
  const float* Wg = (const float*)d_in[9];
  const float* bg = (const float*)d_in[10];
  const float* Wp = (const float*)d_in[11];
  const float* bp = (const float*)d_in[12];
  const float* g1 = (const float*)d_in[13];
  const float* b1 = (const float*)d_in[14];
  const float* g2 = (const float*)d_in[15];
  const float* b2 = (const float*)d_in[16];
  const int* srcI = (const int*)d_in[17];
  const int* tgtI = (const int*)d_in[18];
  const int C = in_sizes[17];
  float* out = (float*)d_out;

  // workspace carve-up
  char* w = (char*)d_ws;
  half_t* x16 = (half_t*)w;            w += (size_t)T_TF * B_BATCH * D_DIM * 2;
  half_t* WvoT = (half_t*)w;           w += (size_t)C * D_DIM * D_DIM * 2;
  float*  bvo = (float*)w;             w += (size_t)C * D_DIM * 4;
  half_t* WgT = (half_t*)w;            w += (size_t)T_TF * 2 * D_DIM * D_DIM * 2;
  half_t* WpT = (half_t*)w;            w += (size_t)T_TF * D_DIM * D_DIM * 2;

  const int ntot = T_TF * B_BATCH * D_DIM;
  k_prep_x<<<ntot / (256 * 4), 256, 0, stream>>>(x, x16, ntot);
  k_prep_wvo<<<C, 256, 0, stream>>>(Wv, Wo, bv, bo, WvoT, bvo);
  k_prep_wgwp<<<T_TF, 256, 0, stream>>>(Wg, Wp, WgT, WpT);

  dim3 grid(B_BATCH / TM, T_TF);
  k_fused<<<grid, 256, 0, stream>>>(x, x16, WvoT, bvo, WgT, bg, WpT, bp,
                                    g1, b1, g2, b2, srcI, tgtI, C, out);
}
